// MultiHeadAttention_41077067219365
// MI455X (gfx1250) — matrix operand order repaired and checked by reading, not compiled, not run on hardware
//
#include <hip/hip_runtime.h>
#include <hip/hip_bf16.h>

// ============================================================================
// MI455X (gfx1250) Multi-Head Attention, bf16 WMMA pipeline, v4.
//
// ~137 GFLOP vs ~150 MB HBM -> compute bound; all six matmul stages run on
// v_wmma_f32_16x16x32_bf16 fed by contiguous ds_load_b128 fragment reads.
// v4 additions:
//   * GEMM tiles staged with global_load_async_to_lds_b128 (ASYNCcnt DMA)
//   * double-buffered LDS in GEMM and attention: async-load tile t+1 while
//     WMMA consumes tile t; s_wait_asynccnt only after compute
//   * softmax reductions on DPP16 (update_dpp) instead of ds_bpermute
// ============================================================================

typedef __bf16 bf16;
typedef bf16  v16bf __attribute__((ext_vector_type(16)));
typedef float v8f   __attribute__((ext_vector_type(8)));
typedef int   v4i   __attribute__((ext_vector_type(4)));

#define D_MODEL 1024
#define SEQ     2048
#define NHEAD   16
#define DK      64

#if __has_builtin(__builtin_amdgcn_global_load_async_to_lds_b128) && \
    __has_builtin(__builtin_amdgcn_s_wait_asynccnt)
#define HAVE_ASYNC 1
// Builtin signature (from hipcc diagnostic): (v4i AS1*, v4i AS3*, Imm, Imm).
#define GLB(p) ((__attribute__((address_space(1))) v4i*)(p))
#define LDS(p) ((__attribute__((address_space(3))) v4i*)(p))
#endif

// fp32 -> bf16 RNE via bit ops (no dependence on __bf16 conversion lowering).
__device__ __forceinline__ unsigned short bfb(float f) {
    unsigned u = __builtin_bit_cast(unsigned, f);
    u += 0x7fffu + ((u >> 16) & 1u);
    return (unsigned short)(u >> 16);
}
__device__ __forceinline__ bf16 f2bf(float f) {
    return __builtin_bit_cast(bf16, bfb(f));
}
__device__ __forceinline__ unsigned pk2(float a, float b) {
    return (unsigned)bfb(a) | ((unsigned)bfb(b) << 16);
}

// ---------------------------------------------------------------------------
// 16-lane butterfly reductions on DPP16 (single-cycle VALU, no LDS traffic).
// quad_perm xor1 (0xB1), quad_perm xor2 (0x4E), row_half_mirror (^7, 0x141),
// row_mirror (^15, 0x140): crosses quads then octets -> full 16-lane result.
// ---------------------------------------------------------------------------
template <int CTRL>
__device__ __forceinline__ float dppf(float x) {
    return __builtin_bit_cast(float, __builtin_amdgcn_update_dpp(
        0, __builtin_bit_cast(int, x), CTRL, 0xF, 0xF, true));
}
__device__ __forceinline__ float red16_max(float x) {
    x = fmaxf(x, dppf<0xB1>(x));
    x = fmaxf(x, dppf<0x4E>(x));
    x = fmaxf(x, dppf<0x141>(x));
    x = fmaxf(x, dppf<0x140>(x));
    return x;
}
__device__ __forceinline__ float red16_sum(float x) {
    x += dppf<0xB1>(x);
    x += dppf<0x4E>(x);
    x += dppf<0x141>(x);
    x += dppf<0x140>(x);
    return x;
}

// ---------------------------------------------------------------------------
// WMMA fragment gathers (CDNA5 ISA 7.12.2, wave32). Both are contiguous
// per-lane (2 x 16B each) -> lower to ds_load_b128.
__device__ __forceinline__ v16bf load_a_frag(const bf16* a, int lda) {
    const int lane  = threadIdx.x & 31;
    const int m     = lane & 15;
    const int khalf = lane >> 4;
    v16bf f;
#pragma unroll
    for (int j = 0; j < 8; ++j) {
        const int k0 = (khalf << 4) + (j << 1);
        f[2 * j]     = a[m * lda + k0];
        f[2 * j + 1] = a[m * lda + k0 + 1];
    }
    return f;
}
__device__ __forceinline__ v16bf load_bt_frag(const bf16* bt, int ldt) {
    const int lane  = threadIdx.x & 31;
    const int n     = lane & 15;
    const int khalf = lane >> 4;
    v16bf f;
#pragma unroll
    for (int j = 0; j < 8; ++j) {
        const int k0 = (khalf << 4) + (j << 1);
        f[2 * j]     = bt[n * ldt + k0];
        f[2 * j + 1] = bt[n * ldt + k0 + 1];
    }
    return f;
}

__device__ __forceinline__ v8f wmma_bf16(v16bf a, v16bf b, v8f c) {
    return __builtin_amdgcn_wmma_f32_16x16x32_bf16(
        false, a, false, b, (short)0, c, false, false);
}

// 16B global->LDS copy: async DMA when available, else load+store.
__device__ __forceinline__ void cp16(const bf16* g, bf16* l) {
#ifdef HAVE_ASYNC
    __builtin_amdgcn_global_load_async_to_lds_b128(GLB(g), LDS(l), 0, 0);
#else
    *(uint4*)l = *(const uint4*)g;
#endif
}
__device__ __forceinline__ void cp_wait() {
#ifdef HAVE_ASYNC
    __builtin_amdgcn_s_wait_asynccnt(0);
#endif
}

// ---------------------------------------------------------------------------
// One-time: fp32 W[K][N] -> bf16 Wt[N][K].
// ---------------------------------------------------------------------------
__global__ __launch_bounds__(256)
void transpose_w_bf16(const float* __restrict__ W, bf16* __restrict__ Wt,
                      int K, int N) {
    __shared__ bf16 t[64][65];
    const int tid = threadIdx.x;
    const int k0 = blockIdx.y * 64, n0 = blockIdx.x * 64;
#pragma unroll
    for (int i = 0; i < 4; ++i) {
        const int slot = i * 256 + tid;            // 1024 float4 slots
        const int r = slot >> 4, c4 = (slot & 15) << 2;
        const float4 f = *(const float4*)&W[(size_t)(k0 + r) * N + n0 + c4];
        t[r][c4 + 0] = f2bf(f.x);
        t[r][c4 + 1] = f2bf(f.y);
        t[r][c4 + 2] = f2bf(f.z);
        t[r][c4 + 3] = f2bf(f.w);
    }
    __syncthreads();
#pragma unroll
    for (int i = 0; i < 2; ++i) {
        const int slot = i * 256 + tid;            // 512 x (8 bf16) slots
        const int r = slot >> 3, c8 = (slot & 7) << 3;
        uint4 o;
        unsigned w[4];
#pragma unroll
        for (int j = 0; j < 4; ++j) {
            const unsigned lo = __builtin_bit_cast(unsigned short, t[c8 + 2 * j][r]);
            const unsigned hi = __builtin_bit_cast(unsigned short, t[c8 + 2 * j + 1][r]);
            w[j] = lo | (hi << 16);
        }
        o.x = w[0]; o.y = w[1]; o.z = w[2]; o.w = w[3];
        *(uint4*)&Wt[(size_t)(n0 + r) * K + k0 + c8] = o;
    }
}

// ---------------------------------------------------------------------------
// One-time per activation: fp32 -> bf16, vectorized.
// ---------------------------------------------------------------------------
__global__ __launch_bounds__(256)
void cvt_f32_to_bf16(const float* __restrict__ x, bf16* __restrict__ y, int n4) {
    const int i = blockIdx.x * 256 + threadIdx.x;
    if (i >= n4) return;
    const float4 f = *(const float4*)&x[(size_t)i * 4];
    uint2 o;
    o.x = pk2(f.x, f.y);
    o.y = pk2(f.z, f.w);
    *(uint2*)&y[(size_t)i * 4] = o;
}

// ---------------------------------------------------------------------------
// GEMM: C = (A[M x K] @ Wt^T + bias) * scale, A bf16 row-major, Wt bf16 [N][K].
// Block tile 128x64, 256 threads = 8 waves, wave = 16 rows x 64 cols
// (4 C fragments), BK = 64. Double-buffered async staging.
// TR_OUT: store C transposed as bf16 [N][M] (vectorized 16B stores).
// ---------------------------------------------------------------------------
template <typename OutT, bool TR_OUT>
__global__ __launch_bounds__(256)
void gemm_bias(const bf16* __restrict__ A, const bf16* __restrict__ Wt,
               const float* __restrict__ bias, OutT* __restrict__ C,
               int M, int N, int K, float scale) {
    __shared__ bf16 sA[2][128][72];
    __shared__ bf16 sBt[2][64][72];

    const int tid  = threadIdx.x;
    const int wid  = tid >> 5;
    const int lane = tid & 31;
    const int ln   = lane & 15;
    const int lh   = lane >> 4;
    const int row0 = blockIdx.y * 128;
    const int col0 = blockIdx.x * 64;

    auto stage = [&](int kt, int buf) {
#pragma unroll
        for (int i = 0; i < 4; ++i) {              // A tile: 1024 x (8 bf16)
            const int slot = i * 256 + tid;
            const int r = slot >> 3, c8 = (slot & 7) << 3;
            cp16(&A[(size_t)(row0 + r) * K + kt + c8], &sA[buf][r][c8]);
        }
#pragma unroll
        for (int i = 0; i < 2; ++i) {              // Wt tile: 512 x (8 bf16)
            const int slot = i * 256 + tid;
            const int r = slot >> 3, c8 = (slot & 7) << 3;
            cp16(&Wt[(size_t)(col0 + r) * K + kt + c8], &sBt[buf][r][c8]);
        }
    };

    v8f acc[4] = {};

    stage(0, 0);
    cp_wait();
    __syncthreads();

    int buf = 0;
    for (int kt = 0; kt < K; kt += 64, buf ^= 1) {
        if (kt + 64 < K) stage(kt + 64, buf ^ 1);  // DMA next tile during WMMA
#pragma unroll
        for (int s = 0; s < 2; ++s) {
            const v16bf a = load_a_frag(&sA[buf][wid * 16][s * 32], 72);
#pragma unroll
            for (int n = 0; n < 4; ++n) {
                const v16bf b = load_bt_frag(&sBt[buf][n * 16][s * 32], 72);
                acc[n] = wmma_bf16(a, b, acc[n]);
            }
        }
        cp_wait();
        __syncthreads();
    }

#pragma unroll
    for (int n = 0; n < 4; ++n) {
        const int col = col0 + n * 16 + ln;
        const float bv = bias[col];
        if constexpr (TR_OUT) {
            // bf16 C^T[N][M]; 8 consecutive rows per lane -> one 16B store.
            uint4 o;
            o.x = pk2((acc[n][0] + bv) * scale, (acc[n][1] + bv) * scale);
            o.y = pk2((acc[n][2] + bv) * scale, (acc[n][3] + bv) * scale);
            o.z = pk2((acc[n][4] + bv) * scale, (acc[n][5] + bv) * scale);
            o.w = pk2((acc[n][6] + bv) * scale, (acc[n][7] + bv) * scale);
            *(uint4*)&C[(size_t)col * M + row0 + wid * 16 + lh * 8] = o;
        } else {
#pragma unroll
            for (int r = 0; r < 8; ++r) {
                const int row = row0 + wid * 16 + lh * 8 + r;
                const float val = (acc[n][r] + bv) * scale;
                if constexpr (sizeof(OutT) == 2)
                    C[(size_t)row * N + col] = f2bf(val);
                else
                    C[(size_t)row * N + col] = val;
            }
        }
    }
}

// ---------------------------------------------------------------------------
// Flash attention. Q, K: bf16 [M][D_MODEL] (head h = cols h*64..+63, Q
// pre-scaled by 1/sqrt(d_k)). Vt: bf16 [D_MODEL][M] (transposed). Grid:
// x = S/64 query blocks, y = B*H. 128 threads = 4 waves x 16 query rows.
// Double-buffered async K/V staging.
// ---------------------------------------------------------------------------
__global__ __launch_bounds__(128)
void attn_kernel(const bf16* __restrict__ Q, const bf16* __restrict__ Kg,
                 const bf16* __restrict__ Vt, bf16* __restrict__ O, int Mtot) {
    __shared__ bf16 sK[2][64][72];   // [key][d] (bt-read for Q*K^T)
    __shared__ bf16 sV[2][64][72];   // [d][key] (bt-read for P*V)
    __shared__ bf16 sP[4][16][72];

    const int tid  = threadIdx.x;
    const int wid  = tid >> 5;
    const int lane = tid & 31;
    const int ln   = lane & 15;
    const int lh   = lane >> 4;

    const int h    = blockIdx.y & (NHEAD - 1);
    const int b    = blockIdx.y >> 4;
    const int col0 = h * DK;
    const int qrow = blockIdx.x * 64 + wid * 16;

    auto stage = [&](int kt, int buf) {
#pragma unroll
        for (int i = 0; i < 4; ++i) {              // 512 x (8 bf16) per tile
            const int slot = i * 128 + tid;
            const int r = slot >> 3, c8 = (slot & 7) << 3;
            cp16(&Kg[(size_t)(b * SEQ + kt + r) * D_MODEL + col0 + c8],
                 &sK[buf][r][c8]);
            cp16(&Vt[(size_t)(col0 + r) * Mtot + b * SEQ + kt + c8],
                 &sV[buf][r][c8]);
        }
    };

    const bf16* qbase = Q + (size_t)(b * SEQ + qrow) * D_MODEL + col0;
    v16bf qf[2];
    qf[0] = load_a_frag(qbase, D_MODEL);
    qf[1] = load_a_frag(qbase + 32, D_MODEL);

    float m_run[8], l_run[8];
#pragma unroll
    for (int r = 0; r < 8; ++r) { m_run[r] = -1e30f; l_run[r] = 0.0f; }
    v8f o_acc[4] = {};

    stage(0, 0);
    cp_wait();
    __syncthreads();

    int buf = 0;
    for (int kt = 0; kt < SEQ; kt += 64, buf ^= 1) {
        if (kt + 64 < SEQ) stage(kt + 64, buf ^ 1);  // DMA next K/V tile

        // ---- scores S = Q * K^T (scale folded into Q) ----
        v8f sc[4] = {};
#pragma unroll
        for (int s = 0; s < 2; ++s) {
#pragma unroll
            for (int n = 0; n < 4; ++n) {
                const v16bf bt = load_bt_frag(&sK[buf][n * 16][s * 32], 72);
                sc[n] = wmma_bf16(qf[s], bt, sc[n]);
            }
        }

        // ---- online softmax (row stats live in one 16-lane half) ----
        float m_new[8], alpha[8];
#pragma unroll
        for (int r = 0; r < 8; ++r) {
            const float mx = red16_max(fmaxf(fmaxf(sc[0][r], sc[1][r]),
                                             fmaxf(sc[2][r], sc[3][r])));
            m_new[r] = fmaxf(m_run[r], mx);
            alpha[r] = __expf(m_run[r] - m_new[r]);
        }
#pragma unroll
        for (int r = 0; r < 8; ++r) {
            float rs = 0.0f;
#pragma unroll
            for (int n = 0; n < 4; ++n) {
                const float p = __expf(sc[n][r] - m_new[r]);
                sc[n][r] = p;
                rs += p;
                o_acc[n][r] *= alpha[r];
            }
            l_run[r] = l_run[r] * alpha[r] + red16_sum(rs);
            m_run[r] = m_new[r];
        }

        // ---- relayout P (C-frag -> per-wave LDS -> A-frag) ----
#pragma unroll
        for (int n = 0; n < 4; ++n)
#pragma unroll
            for (int r = 0; r < 8; ++r)
                sP[wid][lh * 8 + r][n * 16 + ln] = f2bf(sc[n][r]);

        // ---- O += P * V (V^T tile read contiguously) ----
#pragma unroll
        for (int s = 0; s < 2; ++s) {
            const v16bf pa = load_a_frag(&sP[wid][0][s * 32], 72);
#pragma unroll
            for (int n = 0; n < 4; ++n) {
                const v16bf vb = load_bt_frag(&sV[buf][n * 16][s * 32], 72);
                o_acc[n] = wmma_bf16(pa, vb, o_acc[n]);
            }
        }

        cp_wait();
        __syncthreads();
    }

    bf16* obase = O + (size_t)(b * SEQ + qrow) * D_MODEL + col0;
#pragma unroll
    for (int n = 0; n < 4; ++n)
#pragma unroll
        for (int r = 0; r < 8; ++r) {
            const float val = o_acc[n][r] / l_run[r];
            obase[(size_t)(lh * 8 + r) * D_MODEL + n * 16 + ln] = f2bf(val);
        }
}

// ---------------------------------------------------------------------------
extern "C" void kernel_launch(void* const* d_in, const int* in_sizes, int n_in,
                              void* d_out, int out_size, void* d_ws, size_t ws_size,
                              hipStream_t stream) {
    const float* q  = (const float*)d_in[0];
    const float* k  = (const float*)d_in[1];
    const float* v  = (const float*)d_in[2];
    const float* Wq = (const float*)d_in[3];
    const float* bq = (const float*)d_in[4];
    const float* Wk = (const float*)d_in[5];
    const float* bk = (const float*)d_in[6];
    const float* Wv = (const float*)d_in[7];
    const float* bv = (const float*)d_in[8];
    const float* Wo = (const float*)d_in[9];
    const float* bo = (const float*)d_in[10];
    float* out = (float*)d_out;

    const int M = in_sizes[0] / D_MODEL;   // B * S (= 8192)
    const int B = M / SEQ;
    const size_t DD = (size_t)D_MODEL * D_MODEL;
    const size_t MD = (size_t)M * D_MODEL;

    // Workspace: 4 transposed weights (8 MB) + X (16 MB, reused) + Q/K/Vt/attn
    // (64 MB) = 88 MB of bf16.
    bf16* Wqt = (bf16*)d_ws;
    bf16* Wkt = Wqt + DD;
    bf16* Wvt = Wkt + DD;
    bf16* Wot = Wvt + DD;
    bf16* Xb  = Wot + DD;
    bf16* Qb  = Xb + MD;
    bf16* Kb  = Qb + MD;
    bf16* Vtb = Kb + MD;
    bf16* Ab  = Vtb + MD;

    const dim3 tgrd(D_MODEL / 64, D_MODEL / 64);
    transpose_w_bf16<<<tgrd, 256, 0, stream>>>(Wq, Wqt, D_MODEL, D_MODEL);
    transpose_w_bf16<<<tgrd, 256, 0, stream>>>(Wk, Wkt, D_MODEL, D_MODEL);
    transpose_w_bf16<<<tgrd, 256, 0, stream>>>(Wv, Wvt, D_MODEL, D_MODEL);
    transpose_w_bf16<<<tgrd, 256, 0, stream>>>(Wo, Wot, D_MODEL, D_MODEL);

    const int n4 = (int)(MD / 4);
    const dim3 cgrd((n4 + 255) / 256);
    const dim3 ggrd(D_MODEL / 64, M / 128);
    const float qscale = 0.125f;  // 1/sqrt(d_k) folded into Q projection

    // Stream ordering makes Xb safely reusable across the three projections.
    cvt_f32_to_bf16<<<cgrd, 256, 0, stream>>>(q, Xb, n4);
    gemm_bias<bf16, false><<<ggrd, 256, 0, stream>>>(Xb, Wqt, bq, Qb, M, D_MODEL, D_MODEL, qscale);
    cvt_f32_to_bf16<<<cgrd, 256, 0, stream>>>(k, Xb, n4);
    gemm_bias<bf16, false><<<ggrd, 256, 0, stream>>>(Xb, Wkt, bk, Kb, M, D_MODEL, D_MODEL, 1.0f);
    cvt_f32_to_bf16<<<cgrd, 256, 0, stream>>>(v, Xb, n4);
    gemm_bias<bf16, true><<<ggrd, 256, 0, stream>>>(Xb, Wvt, bv, Vtb, M, D_MODEL, D_MODEL, 1.0f);

    const dim3 agrd(SEQ / 64, B * NHEAD);
    attn_kernel<<<agrd, 128, 0, stream>>>(Qb, Kb, Vtb, Ab, M);

    gemm_bias<float, false><<<ggrd, 256, 0, stream>>>(Ab, Wot, bo, out, M, D_MODEL, D_MODEL, 1.0f);
}
